// HyenaOperator_87230785781945
// MI455X (gfx1250) — compile-verified, matrix operand order checked by reading, not run on hardware
//
#include <hip/hip_runtime.h>
#include <hip/hip_bf16.h>

typedef __attribute__((ext_vector_type(16))) __bf16 v16bf;
typedef __attribute__((ext_vector_type(8)))  __bf16 v8bf;
typedef __attribute__((ext_vector_type(8)))  float  v8f;
typedef __attribute__((ext_vector_type(4)))  int    v4i;

#define BB 4
#define SS 2048
#define DD 2048
#define DI 1024
#define FO 2
#define KK 3

// ---- CDNA5 async global->LDS path (guarded: falls back to register staging) ----
#if defined(__HIP_DEVICE_COMPILE__) && __has_builtin(__builtin_amdgcn_global_load_async_to_lds_b128)
#define HAS_ASYNC_LDS 1
#else
#define HAS_ASYNC_LDS 0
#endif

#if HAS_ASYNC_LDS
typedef v4i __attribute__((address_space(1)))* gv4i_p;   // global 128-bit chunk ptr
typedef v4i __attribute__((address_space(3)))* lv4i_p;   // LDS 128-bit chunk ptr

__device__ __forceinline__ void async_cp_b128(const __bf16* g, __bf16* l) {
    __builtin_amdgcn_global_load_async_to_lds_b128((gv4i_p)g, (lv4i_p)l, 0, 0);
}
__device__ __forceinline__ void wait_async0() {
#if __has_builtin(__builtin_amdgcn_s_wait_asynccnt)
    __builtin_amdgcn_s_wait_asynccnt(0);
#else
    asm volatile("s_wait_asynccnt 0" ::: "memory");
#endif
}
#endif

__device__ __forceinline__ float gelu_exact(float v) {
    return 0.5f * v * (1.0f + erff(v * 0.70710678118654752f));
}

// Fragment-swizzled operand layout: tiles of 16 rows x 32 k, stored as
// [rowTile][kTile][lane][16 elems] (1KB per tile, contiguous).
// row = M for the A operand, N for the B operand.  KT = K/32.
__device__ __forceinline__ size_t swz_idx(int row, int k, int KT) {
    int rt = row >> 4, kt = k >> 5, r = row & 15, kk = k & 31;
    int g = (kk >> 3) & 1;
    int e = (kk & 7) + ((kk >> 4) << 3);
    int lane = r + (g << 4);
    return ((((size_t)rt * KT) + kt) << 9) + ((size_t)(lane << 4) + e);
}

// ---------------- fp32 weights -> bf16, fragment-swizzled (B operand) ----------------
__global__ __launch_bounds__(256) void cvt_swz_kernel(const float* __restrict__ src,
                                                      __bf16* __restrict__ dst,
                                                      int Kdim, int Ndim) {
    int i4 = (blockIdx.x * 256 + threadIdx.x) * 4;
    if (i4 >= Kdim * Ndim) return;
    int k = i4 / Ndim;
    int n0 = i4 % Ndim;
    float4 f = *(const float4*)(src + i4);
    int KT = Kdim >> 5;
    dst[swz_idx(n0 + 0, k, KT)] = (__bf16)f.x;
    dst[swz_idx(n0 + 1, k, KT)] = (__bf16)f.y;
    dst[swz_idx(n0 + 2, k, KT)] = (__bf16)f.z;
    dst[swz_idx(n0 + 3, k, KT)] = (__bf16)f.w;
}

// ---------------- causal depthwise short conv + residual -> bf16 (A-swizzled) --------
__global__ __launch_bounds__(256) void shortconv_kernel(const float* __restrict__ x,
                                                        const float* __restrict__ sw,
                                                        const float* __restrict__ sb,
                                                        __bf16* __restrict__ xs) {
    int idx = blockIdx.x * 256 + threadIdx.x;
    int d = idx % DD;
    int m = idx / DD;          // m = b*S + s
    int s = m % SS;
    size_t base = (size_t)idx;
    float x2 = x[base];
    float x1 = (s >= 1) ? x[base - DD] : 0.0f;
    float x0 = (s >= 2) ? x[base - 2 * DD] : 0.0f;
    float v = x2 + sb[d] + x0 * sw[d * 3 + 0] + x1 * sw[d * 3 + 1] + x2 * sw[d * 3 + 2];
    xs[swz_idx(m, d, DD >> 5)] = (__bf16)v;
}

// ---------------- tiled WMMA bf16 GEMM on pre-swizzled operands ----------------
// C = act(A[MxK] @ B[KxN] + bias).  ACT==1: gelu -> bf16 out.  ACT==0: none -> fp32 out.
// Block tile 128x128, BK=32, double-buffered LDS, 8 waves each owning 32x64 (8 WMMA/stage).
template <int ACT>
__global__ __launch_bounds__(256) void gemm_wmma_kernel(const __bf16* __restrict__ As,
                                                        const __bf16* __restrict__ Bs,
                                                        const float* __restrict__ bias,
                                                        void* __restrict__ Cout,
                                                        int M, int N, int K) {
    __shared__ alignas(32) __bf16 sA[2][8 * 512];   // 8 m-tiles x 1KB per stage
    __shared__ alignas(32) __bf16 sB[2][8 * 512];   // 8 n-tiles x 1KB per stage

    const int tid = threadIdx.x;
    const int lane = tid & 31;
    const int wave = tid >> 5;
    const int laneRow = lane & 15;
    const int laneHi = lane >> 4;
    const int KT = K >> 5;

    const int bmT = blockIdx.y * 8;       // base m-tile (16-row tiles)
    const int bnT = blockIdx.x * 8;       // base n-tile (16-col tiles)
    const int wmW = wave & 3;             // wave m-group: 2 m-tiles
    const int wnW = wave >> 2;            // wave n-group: 4 n-tiles

    // staging: 256 threads, 8 tiles per operand, 32 threads/tile, 32B/thread
    const int cpTile = tid >> 5;
    const int cpLane = tid & 31;
    const size_t aRowBase = ((size_t)(bmT + cpTile)) * KT;
    const size_t bRowBase = ((size_t)(bnT + cpTile)) * KT;
    const int cpOff = cpTile * 512 + cpLane * 16;

    v8f acc[2][4];
#pragma unroll
    for (int a = 0; a < 2; ++a)
#pragma unroll
        for (int b = 0; b < 4; ++b) acc[a][b] = (v8f){};

#if HAS_ASYNC_LDS
    // preload stage 0 via async DMA
    {
        const __bf16* gA = As + ((aRowBase + 0) << 9) + cpLane * 16;
        const __bf16* gB = Bs + ((bRowBase + 0) << 9) + cpLane * 16;
        async_cp_b128(gA + 0, &sA[0][cpOff + 0]);
        async_cp_b128(gA + 8, &sA[0][cpOff + 8]);
        async_cp_b128(gB + 0, &sB[0][cpOff + 0]);
        async_cp_b128(gB + 8, &sB[0][cpOff + 8]);
        wait_async0();
    }
    __syncthreads();
#else
    // preload stage 0 via register staging
    {
        v16bf ra = *(const v16bf*)(As + ((aRowBase + 0) << 9) + cpLane * 16);
        v16bf rb = *(const v16bf*)(Bs + ((bRowBase + 0) << 9) + cpLane * 16);
        *(v16bf*)(&sA[0][cpOff]) = ra;
        *(v16bf*)(&sB[0][cpOff]) = rb;
    }
    __syncthreads();
#endif

    for (int kt = 0; kt < KT; ++kt) {
        const int cur = kt & 1;
        const int nxt = cur ^ 1;
        const bool more = (kt + 1) < KT;

#if HAS_ASYNC_LDS
        if (more) {   // fire next-stage DMA before the matrix work
            const __bf16* gA = As + ((aRowBase + kt + 1) << 9) + cpLane * 16;
            const __bf16* gB = Bs + ((bRowBase + kt + 1) << 9) + cpLane * 16;
            async_cp_b128(gA + 0, &sA[nxt][cpOff + 0]);
            async_cp_b128(gA + 8, &sA[nxt][cpOff + 8]);
            async_cp_b128(gB + 0, &sB[nxt][cpOff + 0]);
            async_cp_b128(gB + 8, &sB[nxt][cpOff + 8]);
        }
#else
        v16bf na, nb;
        if (more) {   // issue next-stage global loads before the matrix work
            na = *(const v16bf*)(As + ((aRowBase + kt + 1) << 9) + cpLane * 16);
            nb = *(const v16bf*)(Bs + ((bRowBase + kt + 1) << 9) + cpLane * 16);
        }
#endif

        v16bf af[2], bfr[4];
#pragma unroll
        for (int tm = 0; tm < 2; ++tm)
            af[tm] = *(const v16bf*)(&sA[cur][(wmW * 2 + tm) * 512 + lane * 16]);
#pragma unroll
        for (int tn = 0; tn < 4; ++tn)
            bfr[tn] = *(const v16bf*)(&sB[cur][(wnW * 4 + tn) * 512 + lane * 16]);

#pragma unroll
        for (int tm = 0; tm < 2; ++tm)
#pragma unroll
            for (int tn = 0; tn < 4; ++tn)
                acc[tm][tn] = __builtin_amdgcn_wmma_f32_16x16x32_bf16(
                    false, af[tm], false, bfr[tn], (short)0, acc[tm][tn], false, false);

#if HAS_ASYNC_LDS
        if (more) wait_async0();
#else
        if (more) {
            *(v16bf*)(&sA[nxt][cpOff]) = na;
            *(v16bf*)(&sB[nxt][cpOff]) = nb;
        }
#endif
        __syncthreads();
    }

    // --- epilogue ---
#pragma unroll
    for (int tm = 0; tm < 2; ++tm)
#pragma unroll
        for (int tn = 0; tn < 4; ++tn) {
            int n = (bnT + wnW * 4 + tn) * 16 + laneRow;
            float bv = bias[n];
#pragma unroll
            for (int i = 0; i < 8; ++i) {
                int m = (bmT + wmW * 2 + tm) * 16 + laneHi * 8 + i;
                float v = acc[tm][tn][i] + bv;
                if (ACT == 1)
                    ((__bf16*)Cout)[(size_t)m * N + n] = (__bf16)gelu_exact(v);
                else
                    ((float*)Cout)[(size_t)m * N + n] = v;
            }
        }
}

// ---------------- zero the 12-float w accumulator (in-graph each launch) ----------------
__global__ void zero_wacc_kernel(float* wacc) {
    if (threadIdx.x < BB * KK) wacc[threadIdx.x] = 0.0f;
}

// -------- filter reduce: w[b,k] = 1/S * sum_s sum_f tanh(h[b,s,f,:]@fw[:,k] + fb[k]) ----
__global__ __launch_bounds__(256) void filt_reduce_kernel(const __bf16* __restrict__ h,
                                                          const float* __restrict__ fw,
                                                          const float* __restrict__ fb,
                                                          float* __restrict__ wacc) {
    __shared__ float sred[6][256];
    const int row = blockIdx.x;  // b*S + s
    const int tid = threadIdx.x;
    const __bf16* hr = h + (size_t)row * (FO * DI);

    float acc[6] = {0.f, 0.f, 0.f, 0.f, 0.f, 0.f};
#pragma unroll
    for (int f = 0; f < FO; ++f)
        for (int i = tid; i < DI; i += 256) {
            float hv = (float)hr[f * DI + i];
            acc[f * 3 + 0] += hv * fw[i * 3 + 0];
            acc[f * 3 + 1] += hv * fw[i * 3 + 1];
            acc[f * 3 + 2] += hv * fw[i * 3 + 2];
        }
#pragma unroll
    for (int q = 0; q < 6; ++q) sred[q][tid] = acc[q];
    __syncthreads();
    for (int off = 128; off > 0; off >>= 1) {
        if (tid < off) {
#pragma unroll
            for (int q = 0; q < 6; ++q) sred[q][tid] += sred[q][tid + off];
        }
        __syncthreads();
    }
    if (tid == 0) {
        int b = row / SS;
#pragma unroll
        for (int k = 0; k < KK; ++k) {
            float t = tanhf(sred[k][0] + fb[k]) + tanhf(sred[3 + k][0] + fb[k]);
            atomicAdd(&wacc[b * 3 + k], t * (1.0f / (float)SS));
        }
    }
}

// ---- gated conv: g = gelu((xs (*) w[b]) * v), recompute short conv, A-swizzled out ----
__global__ __launch_bounds__(256) void gate_kernel(const float* __restrict__ x,
                                                   const float* __restrict__ sw,
                                                   const float* __restrict__ sb,
                                                   const float* __restrict__ wacc,
                                                   __bf16* __restrict__ g) {
    int idx = blockIdx.x * 256 + threadIdx.x;
    int d = idx % DD;
    int m = idx / DD;
    int s = m % SS;
    int b = m / SS;
    size_t base = (size_t)idx;

    float xv[5];
#pragma unroll
    for (int t = 0; t < 5; ++t) {
        int st = s - 4 + t;
        xv[t] = (st >= 0) ? x[base + (size_t)(t - 4) * DD] : 0.0f;
    }
    float sw0 = sw[d * 3 + 0], sw1 = sw[d * 3 + 1], sw2 = sw[d * 3 + 2];
    float sbd = sb[d];
    float w0 = wacc[b * 3 + 0], w1 = wacc[b * 3 + 1], w2 = wacc[b * 3 + 2];

    float conv = 0.0f;
#pragma unroll
    for (int u = 0; u < 3; ++u) {
        int sp = s - 2 + u;
        if (sp >= 0) {
            float xsu = xv[2 + u] + sbd + xv[u] * sw0 + xv[u + 1] * sw1 + xv[u + 2] * sw2;
            conv += xsu * ((u == 0) ? w0 : (u == 1) ? w1 : w2);
        }
    }
    g[swz_idx(m, d, DD >> 5)] = (__bf16)gelu_exact(conv * xv[4]);  // xv[4] == v
}

// ================================ launcher ================================
extern "C" void kernel_launch(void* const* d_in, const int* in_sizes, int n_in,
                              void* d_out, int out_size, void* d_ws, size_t ws_size,
                              hipStream_t stream) {
    const float* x       = (const float*)d_in[0];
    const float* short_w = (const float*)d_in[1];
    const float* short_b = (const float*)d_in[2];
    const float* proj_w  = (const float*)d_in[3];
    const float* proj_b  = (const float*)d_in[4];
    const float* filt_w  = (const float*)d_in[5];
    const float* filt_b  = (const float*)d_in[6];
    const float* out_w   = (const float*)d_in[7];
    const float* out_b   = (const float*)d_in[8];
    float* out = (float*)d_out;

    const size_t NTOT = (size_t)BB * SS * DD;  // 16,777,216
    char* ws = (char*)d_ws;
    size_t off = 0;
    auto take = [&](size_t bytes) {
        size_t o = off;
        off = (off + bytes + 255) & ~(size_t)255;
        return o;
    };
    __bf16* xs_bf = (__bf16*)(ws + take(NTOT * 2));                  // 32 MB (reused as g)
    __bf16* h_bf  = (__bf16*)(ws + take(NTOT * 2));                  // 32 MB
    __bf16* pw_bf = (__bf16*)(ws + take((size_t)DD * FO * DI * 2));  // 8 MB (swizzled)
    __bf16* ow_bf = (__bf16*)(ws + take((size_t)DD * DD * 2));       // 8 MB (swizzled)
    float*  wacc  = (float*)(ws + take(64));
    __bf16* g_bf  = xs_bf;  // alias: xs dead after GEMM1

    // 1) weights -> bf16, fragment-swizzled
    {
        int n = DD * FO * DI;
        cvt_swz_kernel<<<n / (4 * 256), 256, 0, stream>>>(proj_w, pw_bf, DD, FO * DI);
        int n2 = DD * DD;
        cvt_swz_kernel<<<n2 / (4 * 256), 256, 0, stream>>>(out_w, ow_bf, DD, DD);
    }
    // 2) short conv + residual -> xs (bf16, A-swizzled)
    shortconv_kernel<<<(int)(NTOT / 256), 256, 0, stream>>>(x, short_w, short_b, xs_bf);

    // 3) GEMM1: h = gelu(xs @ proj_w + proj_b)   (8192x2048 @ 2048x2048)
    {
        dim3 grid((FO * DI) / 128, (BB * SS) / 128);
        gemm_wmma_kernel<1><<<grid, 256, 0, stream>>>(xs_bf, pw_bf, proj_b, (void*)h_bf,
                                                      BB * SS, FO * DI, DD);
    }
    // 4) filter reduce -> w[b,k]
    zero_wacc_kernel<<<1, 32, 0, stream>>>(wacc);
    filt_reduce_kernel<<<BB * SS, 256, 0, stream>>>(h_bf, filt_w, filt_b, wacc);

    // 5) gated conv + gelu -> g (bf16, A-swizzled)
    gate_kernel<<<(int)(NTOT / 256), 256, 0, stream>>>(x, short_w, short_b, wacc, g_bf);

    // 6) GEMM2: out = g @ out_w + out_b (fp32 out)
    {
        dim3 grid(DD / 128, (BB * SS) / 128);
        gemm_wmma_kernel<0><<<grid, 256, 0, stream>>>(g_bf, ow_bf, out_b, (void*)out,
                                                      BB * SS, DD, DD);
    }
    (void)in_sizes; (void)n_in; (void)out_size; (void)ws_size;
}
